// DeformableTransformerDecoderLayer_1898375545514
// MI455X (gfx1250) — compile-verified, matrix operand order checked by reading, not run on hardware
//
#include <hip/hip_runtime.h>
#include <hip/hip_bf16.h>

typedef __attribute__((ext_vector_type(16))) _Float16 v16h;
typedef __attribute__((ext_vector_type(8)))  _Float16 v8h;
typedef __attribute__((ext_vector_type(4)))  _Float16 v4h;
typedef __attribute__((ext_vector_type(8)))  float    v8f;
typedef __attribute__((ext_vector_type(4)))  unsigned int v4u;
typedef __attribute__((ext_vector_type(8)))  int      v8i;
typedef __attribute__((ext_vector_type(4)))  int      v4i;

#define D_    256
#define NQ_   900
#define BS_   8
#define NL_   4
#define NH_   8
#define NP_   4
#define DFF_  1024
#define HD_   32
#define T_    (NQ_*BS_)       /* 7200 query tokens */
#define LMEM_ 13294
#define MMEM_ (LMEM_*BS_)     /* 106352 memory tokens */
#define KPAD_ 928             /* keys padded to mult of 32 */

#define GEMM_RELU   1
#define GEMM_OUTF16 2

// ---------------------------------------------------------------------------
// WMMA helpers (wave32, gfx1250)
// A 16x32 f16 layout: lane -> row M=lane&15; elem i -> K = (i>>3)*16 + (lane>>4)*8 + (i&7)
// B 32x16 f16 layout: lane -> col N=lane&15; same K mapping.
// C/D f32 16x16: reg r at lane L holds (M = r + 8*(L>>4), N = L&15).
// ---------------------------------------------------------------------------
__device__ __forceinline__ v8f wmma_f16(v16h a, v16h b, v8f c) {
  return __builtin_amdgcn_wmma_f32_16x16x32_f16(false, a, false, b, (short)0, c,
                                                false, false);
}
__device__ __forceinline__ v16h cat16(v8h lo, v8h hi) {
  v16h r;
#pragma unroll
  for (int i = 0; i < 8; ++i) { r[i] = lo[i]; r[8 + i] = hi[i]; }
  return r;
}

#if __has_builtin(__builtin_amdgcn_tensor_load_to_lds)
#define USE_TDM 1
#else
#define USE_TDM 0
#endif

#if USE_TDM
// ---------------------------------------------------------------------------
// TDM: DMA one 2-D tile of 64 rows x 64 f16 from a row-major [rows x K]
// tensor into LDS, with hardware row padding to a 72-f16 pitch
// (pad_interval code 4 = every 32 DWORDs, pad_amount code 3 = 4 DWORDs).
// rows_rem clamps OOB rows to zero via tensor_dim1 (TDM zero-fills).
// ---------------------------------------------------------------------------
__device__ __forceinline__ void tdm_load_tile(unsigned int lds_addr,
                                              const _Float16* gptr,
                                              int rows_rem, int cols_rem,
                                              int row_stride_elems) {
  unsigned long long ga = (unsigned long long)(uintptr_t)gptr;
  v4u g0;
  g0[0] = 1u;                                         // count=1, user descriptor
  g0[1] = lds_addr;                                   // LDS byte address
  g0[2] = (unsigned int)(ga & 0xffffffffu);           // global_addr[31:0]
  g0[3] = (unsigned int)((ga >> 32) & 0x01ffffffu) | (2u << 30);  // addr[56:32], type=2
  unsigned int td0 = (unsigned int)cols_rem;          // tensor_dim0 (elems)
  unsigned int td1 = (unsigned int)rows_rem;          // tensor_dim1 (rows)
  unsigned int st  = (unsigned int)row_stride_elems;  // tensor_dim0_stride
  v8i g1;
  g1[0] = (int)((1u << 16)      // data_size = 1 -> 2 bytes
              | (1u << 20)      // pad_enable
              | (4u << 22)      // pad_interval: 32 DWORDs (= one 64-f16 row)
              | (3u << 25));    // pad_amount: 4 DWORDs (= 8 f16)
  g1[1] = (int)((td0 & 0xffffu) << 16);               // tensor_dim0[15:0]
  g1[2] = (int)((td0 >> 16) | ((td1 & 0xffffu) << 16));
  g1[3] = (int)((td1 >> 16) | (64u << 16));           // tile_dim0 = 64
  g1[4] = (int)64u;                                   // tile_dim1 = 64, tile_dim2 = 0
  g1[5] = (int)st;                                    // dim0 stride [31:0]
  g1[6] = 0;
  g1[7] = 0;
  v4i zz = {0, 0, 0, 0};
  v8i z8 = {0, 0, 0, 0, 0, 0, 0, 0};
  __builtin_amdgcn_tensor_load_to_lds(g0, g1, zz, zz, z8, 0);
}
#endif

// ---------------------------------------------------------------------------
// Generic GEMM: C[M,N] = A[M,K] @ W[N,K]^T + bias, A/W in f16,
// out f32 or f16 (flags). Block = 128 threads (4 waves), tile 64x64, BK=64.
// LDS tiles are filled by the Tensor Data Mover (wave 0 issues two
// descriptors, waits on TENSORcnt, then the workgroup barrier publishes).
// ---------------------------------------------------------------------------
__global__ __launch_bounds__(128) void gemm_bias(
    const _Float16* __restrict__ A, const _Float16* __restrict__ W,
    const float* __restrict__ bias, void* __restrict__ Cout,
    int M, int N, int K, int flags) {
  __shared__ __align__(16) _Float16 Asb[64][72];
  __shared__ __align__(16) _Float16 Wsb[64][72];
  const int tid  = threadIdx.x;
  const int lane = tid & 31, wid = tid >> 5;
  const int l15  = lane & 15, half = lane >> 4;
  const int n0 = blockIdx.x * 64, m0 = blockIdx.y * 64;

  v8f z = {};
  v8f acc[4];
#pragma unroll
  for (int i = 0; i < 4; ++i) acc[i] = z;

  for (int kb = 0; kb < K; kb += 64) {
#if USE_TDM
    if (tid < 32) {   // wave 0 programs the Tensor Data Mover
      tdm_load_tile((unsigned int)(uintptr_t)&Asb[0][0],
                    A + (size_t)m0 * K + kb, M - m0, K - kb, K);
      tdm_load_tile((unsigned int)(uintptr_t)&Wsb[0][0],
                    W + (size_t)n0 * K + kb, N - n0, K - kb, K);
      __builtin_amdgcn_s_wait_tensorcnt((short)0);
    }
    __syncthreads();
#else
    // fallback: manual staging, 4 x 16B chunks per thread per matrix
    v8h zero8 = {};
#pragma unroll
    for (int j = 0; j < 4; ++j) {
      int chunk = tid + 128 * j;           // 0..511, 8 chunks per row
      int r = chunk >> 3, c8 = (chunk & 7) * 8;
      int gm = m0 + r; int gmc = gm < M ? gm : M - 1;
      int gn = n0 + r; int gnc = gn < N ? gn : N - 1;
      v8h av = *(const v8h*)(A + (size_t)gmc * K + kb + c8);
      v8h wv = *(const v8h*)(W + (size_t)gnc * K + kb + c8);
      if (gm >= M) av = zero8;
      if (gn >= N) wv = zero8;
      *(v8h*)&Asb[r][c8] = av;
      *(v8h*)&Wsb[r][c8] = wv;
    }
    __syncthreads();
#endif

    const int mrow = wid * 16 + l15;
    v16h af0 = cat16(*(const v8h*)&Asb[mrow][half * 8],
                     *(const v8h*)&Asb[mrow][16 + half * 8]);
    v16h af1 = cat16(*(const v8h*)&Asb[mrow][32 + half * 8],
                     *(const v8h*)&Asb[mrow][48 + half * 8]);
#pragma unroll
    for (int nt = 0; nt < 4; ++nt) {
      const int nrow = nt * 16 + l15;
      v16h bf0 = cat16(*(const v8h*)&Wsb[nrow][half * 8],
                       *(const v8h*)&Wsb[nrow][16 + half * 8]);
      v16h bf1 = cat16(*(const v8h*)&Wsb[nrow][32 + half * 8],
                       *(const v8h*)&Wsb[nrow][48 + half * 8]);
      acc[nt] = wmma_f16(af0, bf0, acc[nt]);
      acc[nt] = wmma_f16(af1, bf1, acc[nt]);
    }
    __syncthreads();
  }

  const bool relu = (flags & GEMM_RELU) != 0;
  const bool of16 = (flags & GEMM_OUTF16) != 0;
#pragma unroll
  for (int nt = 0; nt < 4; ++nt) {
#pragma unroll
    for (int r = 0; r < 8; ++r) {
      int gm = m0 + wid * 16 + r + 8 * half;
      int gn = n0 + nt * 16 + l15;
      if (gm < M) {
        float v = acc[nt][r] + bias[gn];
        if (relu) v = fmaxf(v, 0.f);
        if (of16) ((_Float16*)Cout)[(size_t)gm * N + gn] = (_Float16)v;
        else      ((float*)Cout)[(size_t)gm * N + gn] = v;
      }
    }
  }
}

// ---------------------------------------------------------------------------
// f32 -> f16 conversion (8 elems/thread, b128 in / b128 out)
// ---------------------------------------------------------------------------
__global__ void cvt_f16(const float* __restrict__ a, _Float16* __restrict__ o,
                        int n8) {
  int i = blockIdx.x * blockDim.x + threadIdx.x;
  if (i >= n8) return;
  float4 v0 = ((const float4*)a)[2 * i];
  float4 v1 = ((const float4*)a)[2 * i + 1];
  v8h h;
  h[0] = (_Float16)v0.x; h[1] = (_Float16)v0.y;
  h[2] = (_Float16)v0.z; h[3] = (_Float16)v0.w;
  h[4] = (_Float16)v1.x; h[5] = (_Float16)v1.y;
  h[6] = (_Float16)v1.z; h[7] = (_Float16)v1.w;
  ((v8h*)o)[i] = h;
}

// o16 = f16(a + b)
__global__ void add_cvt(const float* __restrict__ a, const float* __restrict__ b,
                        _Float16* __restrict__ o, int n8) {
  int i = blockIdx.x * blockDim.x + threadIdx.x;
  if (i >= n8) return;
  float4 a0 = ((const float4*)a)[2 * i], a1 = ((const float4*)a)[2 * i + 1];
  float4 b0 = ((const float4*)b)[2 * i], b1 = ((const float4*)b)[2 * i + 1];
  v8h h;
  h[0] = (_Float16)(a0.x + b0.x); h[1] = (_Float16)(a0.y + b0.y);
  h[2] = (_Float16)(a0.z + b0.z); h[3] = (_Float16)(a0.w + b0.w);
  h[4] = (_Float16)(a1.x + b1.x); h[5] = (_Float16)(a1.y + b1.y);
  h[6] = (_Float16)(a1.z + b1.z); h[7] = (_Float16)(a1.w + b1.w);
  ((v8h*)o)[i] = h;
}

// ---------------------------------------------------------------------------
// Repack QKV projections into padded f16 per-head layouts.
// Q16/K16: [b,h,KPAD,32] row-major; Vt16: [b,h,32,KPAD] (transposed).
// ---------------------------------------------------------------------------
__global__ void repack_qkv(const float* __restrict__ qk,
                           const float* __restrict__ vp,
                           _Float16* __restrict__ Q16,
                           _Float16* __restrict__ K16,
                           _Float16* __restrict__ Vt16) {
  int idx = blockIdx.x * blockDim.x + threadIdx.x;
  if (idx >= BS_ * NH_ * KPAD_ * HD_) return;
  int d = idx & (HD_ - 1);
  int n = (idx / HD_) % KPAD_;
  int h = (idx / (HD_ * KPAD_)) % NH_;
  int b = idx / (HD_ * KPAD_ * NH_);
  float qv = 0.f, kv = 0.f, vv = 0.f;
  if (n < NQ_) {
    size_t row = (size_t)(n * BS_ + b);
    qv = qk[row * 512 + h * HD_ + d] * 0.17677669529663689f;  // 1/sqrt(32)
    kv = qk[row * 512 + 256 + h * HD_ + d];
    vv = vp[row * 256 + h * HD_ + d];
  }
  size_t base = (size_t)(b * NH_ + h) * KPAD_ * HD_;
  Q16[base + (size_t)n * HD_ + d] = (_Float16)qv;
  K16[base + (size_t)n * HD_ + d] = (_Float16)kv;
  Vt16[((size_t)(b * NH_ + h) * HD_ + d) * KPAD_ + n] = (_Float16)vv;
}

// ---------------------------------------------------------------------------
// Fused self-attention, one wave per 16-query tile per (b,h).
// S^T = K . Q^T  -> per-lane online softmax -> O^T = V^T . P^T where P^T's
// B layout equals S^T's C layout (register chaining, no shuffles).
// Output written directly as f16 (feeds the out_proj GEMM).
// ---------------------------------------------------------------------------
__global__ __launch_bounds__(32) void attn_fused(
    const _Float16* __restrict__ Q16, const _Float16* __restrict__ K16,
    const _Float16* __restrict__ Vt16, _Float16* __restrict__ out) {
  const int qbase = blockIdx.x * 16;
  const int h = blockIdx.y, b = blockIdx.z;
  const int lane = threadIdx.x;
  const int l15 = lane & 15, half = lane >> 4;
  const _Float16* Qp = Q16 + (size_t)(b * NH_ + h) * KPAD_ * HD_;
  const _Float16* Kp = K16 + (size_t)(b * NH_ + h) * KPAD_ * HD_;
  const _Float16* Vp = Vt16 + (size_t)(b * NH_ + h) * HD_ * KPAD_;

  const int q = qbase + l15;
  v16h bq = cat16(*(const v8h*)(Qp + (size_t)q * HD_ + half * 8),
                  *(const v8h*)(Qp + (size_t)q * HD_ + 16 + half * 8));

  float m = -3.0e38f, lsum = 0.f;
  v8f acc0 = {}, acc1 = {};

  for (int kb = 0; kb < KPAD_; kb += 32) {
    v16h a0 = cat16(*(const v8h*)(Kp + (size_t)(kb + l15) * HD_ + half * 8),
                    *(const v8h*)(Kp + (size_t)(kb + l15) * HD_ + 16 + half * 8));
    v16h a1 = cat16(*(const v8h*)(Kp + (size_t)(kb + 16 + l15) * HD_ + half * 8),
                    *(const v8h*)(Kp + (size_t)(kb + 16 + l15) * HD_ + 16 + half * 8));
    v8f z = {};
    v8f s0 = wmma_f16(a0, bq, z);
    v8f s1 = wmma_f16(a1, bq, z);

    float bm = -3.0e38f;
#pragma unroll
    for (int r = 0; r < 8; ++r) {
      if (kb + r + 8 * half >= NQ_)      s0[r] = -1.0e30f;
      if (kb + 16 + r + 8 * half >= NQ_) s1[r] = -1.0e30f;
      bm = fmaxf(bm, fmaxf(s0[r], s1[r]));
    }
    bm = fmaxf(bm, __shfl_xor(bm, 16, 32));
    float mnew = fmaxf(m, bm);
    float corr = __expf(m - mnew);
    m = mnew;

    float p0[8], p1[8], ps = 0.f;
#pragma unroll
    for (int r = 0; r < 8; ++r) {
      p0[r] = __expf(s0[r] - mnew);
      p1[r] = __expf(s1[r] - mnew);
      ps += p0[r] + p1[r];
    }
    ps += __shfl_xor(ps, 16, 32);
    lsum = lsum * corr + ps;

#pragma unroll
    for (int r = 0; r < 8; ++r) { acc0[r] *= corr; acc1[r] *= corr; }

    v16h bp;
#pragma unroll
    for (int i = 0; i < 8; ++i) {
      bp[i]     = (_Float16)p0[i];
      bp[8 + i] = (_Float16)p1[i];
    }

    v16h av0 = cat16(*(const v8h*)(Vp + (size_t)l15 * KPAD_ + kb + half * 8),
                     *(const v8h*)(Vp + (size_t)l15 * KPAD_ + kb + 16 + half * 8));
    v16h av1 = cat16(*(const v8h*)(Vp + (size_t)(16 + l15) * KPAD_ + kb + half * 8),
                     *(const v8h*)(Vp + (size_t)(16 + l15) * KPAD_ + kb + 16 + half * 8));
    acc0 = wmma_f16(av0, bp, acc0);
    acc1 = wmma_f16(av1, bp, acc1);
  }

  float inv = 1.0f / lsum;
  if (q < NQ_) {
    // acc0 regs r=0..7 hold d = 8*half + r (contiguous) -> one b128 store each
    v8h o0, o1;
#pragma unroll
    for (int r = 0; r < 8; ++r) {
      o0[r] = (_Float16)(acc0[r] * inv);
      o1[r] = (_Float16)(acc1[r] * inv);
    }
    size_t row = (size_t)(q * BS_ + b) * D_ + h * HD_;
    *(v8h*)(out + row + 8 * half)      = o0;
    *(v8h*)(out + row + 16 + 8 * half) = o1;
  }
}

// ---------------------------------------------------------------------------
// y = LayerNorm(x + res) * g + b; optional extra f16 copy (GEMM operand)
// ---------------------------------------------------------------------------
__global__ __launch_bounds__(128) void resid_ln(
    const float* __restrict__ x, const float* __restrict__ res,
    const float* __restrict__ g, const float* __restrict__ bt,
    float* __restrict__ y, _Float16* __restrict__ y16, int T) {
  int wid = threadIdx.x >> 5, lane = threadIdx.x & 31;
  int t = blockIdx.x * 4 + wid;
  if (t >= T) return;
  size_t base = (size_t)t * 256 + lane * 8;
  float4 x0 = *(const float4*)(x + base);
  float4 x1 = *(const float4*)(x + base + 4);
  float4 r0 = *(const float4*)(res + base);
  float4 r1 = *(const float4*)(res + base + 4);
  float v[8] = {x0.x + r0.x, x0.y + r0.y, x0.z + r0.z, x0.w + r0.w,
                x1.x + r1.x, x1.y + r1.y, x1.z + r1.z, x1.w + r1.w};
  float s = 0.f;
#pragma unroll
  for (int j = 0; j < 8; ++j) s += v[j];
#pragma unroll
  for (int o = 16; o; o >>= 1) s += __shfl_xor(s, o, 32);
  float mean = s * (1.f / 256.f);
  float var = 0.f;
#pragma unroll
  for (int j = 0; j < 8; ++j) { float d = v[j] - mean; var += d * d; }
#pragma unroll
  for (int o = 16; o; o >>= 1) var += __shfl_xor(var, o, 32);
  float inv = rsqrtf(var * (1.f / 256.f) + 1e-5f);
  float4 g0 = *(const float4*)(g + lane * 8);
  float4 g1 = *(const float4*)(g + lane * 8 + 4);
  float4 b0 = *(const float4*)(bt + lane * 8);
  float4 b1 = *(const float4*)(bt + lane * 8 + 4);
  float o_[8];
  o_[0] = (v[0] - mean) * inv * g0.x + b0.x;
  o_[1] = (v[1] - mean) * inv * g0.y + b0.y;
  o_[2] = (v[2] - mean) * inv * g0.z + b0.z;
  o_[3] = (v[3] - mean) * inv * g0.w + b0.w;
  o_[4] = (v[4] - mean) * inv * g1.x + b1.x;
  o_[5] = (v[5] - mean) * inv * g1.y + b1.y;
  o_[6] = (v[6] - mean) * inv * g1.z + b1.z;
  o_[7] = (v[7] - mean) * inv * g1.w + b1.w;
  float4 w0 = {o_[0], o_[1], o_[2], o_[3]};
  float4 w1 = {o_[4], o_[5], o_[6], o_[7]};
  *(float4*)(y + base)     = w0;
  *(float4*)(y + base + 4) = w1;
  if (y16) {
    v8h hh;
#pragma unroll
    for (int j = 0; j < 8; ++j) hh[j] = (_Float16)o_[j];
    *(v8h*)(y16 + base) = hh;
  }
}

// softmax over the 16 (level,point) weights per (token, head)
__global__ void aw_softmax(float* __restrict__ aw) {
  int i = blockIdx.x * blockDim.x + threadIdx.x;
  if (i >= T_ * NH_) return;
  float* p = aw + (size_t)i * 16;
  float4 r[4];
#pragma unroll
  for (int j = 0; j < 4; ++j) r[j] = *(const float4*)(p + 4 * j);
  float e[16] = {r[0].x, r[0].y, r[0].z, r[0].w, r[1].x, r[1].y, r[1].z, r[1].w,
                 r[2].x, r[2].y, r[2].z, r[2].w, r[3].x, r[3].y, r[3].z, r[3].w};
  float mx = -3.0e38f;
#pragma unroll
  for (int j = 0; j < 16; ++j) mx = fmaxf(mx, e[j]);
  float s = 0.f;
#pragma unroll
  for (int j = 0; j < 16; ++j) { e[j] = __expf(e[j] - mx); s += e[j]; }
  float inv = 1.f / s;
#pragma unroll
  for (int j = 0; j < 16; ++j) e[j] *= inv;
  float4 o[4] = {{e[0], e[1], e[2], e[3]}, {e[4], e[5], e[6], e[7]},
                 {e[8], e[9], e[10], e[11]}, {e[12], e[13], e[14], e[15]}};
#pragma unroll
  for (int j = 0; j < 4; ++j) *(float4*)(p + 4 * j) = o[j];
}

// ---------------------------------------------------------------------------
// Deformable bilinear sampling; writes f16 (feeds output_proj GEMM).
// A wave's 32 lanes span d=0..31 of one (t,h): contiguous 128B value bursts.
// ---------------------------------------------------------------------------
__global__ void deform_attn(const float* __restrict__ off,
                            const float* __restrict__ aw,
                            const float* __restrict__ ref,
                            const float* __restrict__ value,
                            _Float16* __restrict__ out) {
  int idx = blockIdx.x * blockDim.x + threadIdx.x;
  if (idx >= T_ * D_) return;
  int c = idx & 255, t = idx >> 8;
  int h = c >> 5, d = c & 31;
  int b = t % BS_;  // token t = n*BS + b

  const int Hs[4] = {100, 50, 25, 13};
  const int Ws[4] = {100, 50, 25, 13};
  const int Ls[4] = {0, 10000, 12500, 13125};

  float accum = 0.f;
#pragma unroll
  for (int l = 0; l < 4; ++l) {
    const int Hl = Hs[l], Wl = Ws[l], s = Ls[l];
    float rx = ref[((size_t)t * 4 + l) * 2 + 0];
    float ry = ref[((size_t)t * 4 + l) * 2 + 1];
#pragma unroll
    for (int p = 0; p < 4; ++p) {
      int oc = ((h * 4 + l) * 4 + p) * 2;
      float x = (rx + off[(size_t)t * 256 + oc]     / (float)Wl) * Wl - 0.5f;
      float y = (ry + off[(size_t)t * 256 + oc + 1] / (float)Hl) * Hl - 0.5f;
      float w = aw[(size_t)t * 128 + h * 16 + l * 4 + p];
      float x0 = floorf(x), y0 = floorf(y);
#pragma unroll
      for (int dy = 0; dy < 2; ++dy) {
#pragma unroll
        for (int dx = 0; dx < 2; ++dx) {
          float xc = x0 + dx, yc = y0 + dy;
          if (xc >= 0.f && xc < (float)Wl && yc >= 0.f && yc < (float)Hl) {
            float bw = (1.f - fabsf(x - xc)) * (1.f - fabsf(y - yc));
            int li = s + (int)yc * Wl + (int)xc;
            accum += w * bw * value[((size_t)li * BS_ + b) * 256 + h * 32 + d];
          }
        }
      }
    }
  }
  out[(size_t)t * 256 + h * 32 + d] = (_Float16)accum;
}

// ---------------------------------------------------------------------------
extern "C" void kernel_launch(void* const* d_in, const int* in_sizes, int n_in,
                              void* d_out, int out_size, void* d_ws, size_t ws_size,
                              hipStream_t stream) {
  (void)in_sizes; (void)n_in; (void)out_size; (void)ws_size;

  const float* tgt           = (const float*)d_in[0];
  const float* pos           = (const float*)d_in[1];
  const float* ref           = (const float*)d_in[2];
  const float* memory        = (const float*)d_in[3];
  const float* in_proj_w     = (const float*)d_in[6];
  const float* in_proj_b     = (const float*)d_in[7];
  const float* out_proj_w    = (const float*)d_in[8];
  const float* out_proj_b    = (const float*)d_in[9];
  const float* samp_off_w    = (const float*)d_in[10];
  const float* samp_off_b    = (const float*)d_in[11];
  const float* attn_w_w      = (const float*)d_in[12];
  const float* attn_w_b      = (const float*)d_in[13];
  const float* value_proj_w  = (const float*)d_in[14];
  const float* value_proj_b  = (const float*)d_in[15];
  const float* output_proj_w = (const float*)d_in[16];
  const float* output_proj_b = (const float*)d_in[17];
  const float* lin1_w        = (const float*)d_in[18];
  const float* lin1_b        = (const float*)d_in[19];
  const float* lin2_w        = (const float*)d_in[20];
  const float* lin2_b        = (const float*)d_in[21];
  const float* n1g = (const float*)d_in[22];
  const float* n1b = (const float*)d_in[23];
  const float* n2g = (const float*)d_in[24];
  const float* n2b = (const float*)d_in[25];
  const float* n3g = (const float*)d_in[26];
  const float* n3b = (const float*)d_in[27];
  float* outp = (float*)d_out;

  char* base = (char*)d_ws;
  size_t woff = 0;
  auto alloc = [&](size_t bytes) -> void* {
    void* p = base + woff;
    woff += (bytes + 255) & ~(size_t)255;
    return p;
  };

  // f16 weight arena
  _Float16* w_inproj = (_Float16*)alloc((size_t)768 * 256 * 2);
  _Float16* w_outp   = (_Float16*)alloc((size_t)256 * 256 * 2);
  _Float16* w_samp   = (_Float16*)alloc((size_t)256 * 256 * 2);
  _Float16* w_attn   = (_Float16*)alloc((size_t)128 * 256 * 2);
  _Float16* w_val    = (_Float16*)alloc((size_t)256 * 256 * 2);
  _Float16* w_oproj  = (_Float16*)alloc((size_t)256 * 256 * 2);
  _Float16* w_lin1   = (_Float16*)alloc((size_t)1024 * 256 * 2);
  _Float16* w_lin2   = (_Float16*)alloc((size_t)256 * 1024 * 2);
  // activations
  _Float16* qbuf16 = (_Float16*)alloc((size_t)T_ * 256 * 2);
  _Float16* tgt16  = (_Float16*)alloc((size_t)T_ * 256 * 2);
  float*    qkproj = (float*)alloc((size_t)T_ * 512 * 4);
  float*    vproj  = (float*)alloc((size_t)T_ * 256 * 4);
  _Float16* Q16    = (_Float16*)alloc((size_t)BS_ * NH_ * KPAD_ * HD_ * 2);
  _Float16* K16    = (_Float16*)alloc((size_t)BS_ * NH_ * KPAD_ * HD_ * 2);
  _Float16* Vt16   = (_Float16*)alloc((size_t)BS_ * NH_ * KPAD_ * HD_ * 2);
  _Float16* sapre16= (_Float16*)alloc((size_t)T_ * 256 * 2);
  float*    sa     = (float*)alloc((size_t)T_ * 256 * 4);
  float*    t2     = (float*)alloc((size_t)T_ * 256 * 4);
  _Float16* qb16   = (_Float16*)alloc((size_t)T_ * 256 * 2);
  _Float16* mem16  = (_Float16*)alloc((size_t)MMEM_ * 256 * 2);
  float*    value  = (float*)alloc((size_t)MMEM_ * 256 * 4);
  float*    offb   = (float*)alloc((size_t)T_ * 256 * 4);
  float*    awb    = (float*)alloc((size_t)T_ * 128 * 4);
  _Float16* dout16 = (_Float16*)alloc((size_t)T_ * 256 * 2);
  float*    ca     = (float*)alloc((size_t)T_ * 256 * 4);
  float*    t3     = (float*)alloc((size_t)T_ * 256 * 4);
  _Float16* t3_16  = (_Float16*)alloc((size_t)T_ * 256 * 2);
  _Float16* ffh16  = (_Float16*)alloc((size_t)T_ * 1024 * 2);
  float*    ffo    = (float*)alloc((size_t)T_ * 256 * 4);

  auto cvt = [&](const float* a, _Float16* o, size_t n) {
    int n8 = (int)(n / 8);
    cvt_f16<<<(n8 + 255) / 256, 256, 0, stream>>>(a, o, n8);
  };
  auto gemm = [&](const _Float16* A, const _Float16* W, const float* bias,
                  void* Cm, int M, int N, int K, int flags) {
    dim3 g(N / 64, (M + 63) / 64);
    gemm_bias<<<g, 128, 0, stream>>>(A, W, bias, Cm, M, N, K, flags);
  };

  // one-time f16 conversions
  cvt(in_proj_w, w_inproj, (size_t)768 * 256);
  cvt(out_proj_w, w_outp, (size_t)256 * 256);
  cvt(samp_off_w, w_samp, (size_t)256 * 256);
  cvt(attn_w_w, w_attn, (size_t)128 * 256);
  cvt(value_proj_w, w_val, (size_t)256 * 256);
  cvt(output_proj_w, w_oproj, (size_t)256 * 256);
  cvt(lin1_w, w_lin1, (size_t)1024 * 256);
  cvt(lin2_w, w_lin2, (size_t)256 * 1024);
  cvt(tgt, tgt16, (size_t)T_ * 256);
  cvt(memory, mem16, (size_t)MMEM_ * 256);

  // --- self attention ---
  add_cvt<<<(T_ * 32 + 255) / 256, 256, 0, stream>>>(tgt, pos, qbuf16, T_ * 32);
  gemm(qbuf16, w_inproj, in_proj_b, qkproj, T_, 512, 256, 0);             // Q|K
  gemm(tgt16, w_inproj + 512 * 256, in_proj_b + 512, vproj, T_, 256, 256, 0);
  {
    int tot = BS_ * NH_ * KPAD_ * HD_;
    repack_qkv<<<(tot + 255) / 256, 256, 0, stream>>>(qkproj, vproj, Q16, K16, Vt16);
  }
  attn_fused<<<dim3((NQ_ + 15) / 16, NH_, BS_), 32, 0, stream>>>(Q16, K16, Vt16, sapre16);
  gemm(sapre16, w_outp, out_proj_b, sa, T_, 256, 256, 0);
  resid_ln<<<(T_ + 3) / 4, 128, 0, stream>>>(tgt, sa, n2g, n2b, t2,
                                             (_Float16*)nullptr, T_);

  // --- deformable cross attention ---
  add_cvt<<<(T_ * 32 + 255) / 256, 256, 0, stream>>>(t2, pos, qb16, T_ * 32);
  gemm(mem16, w_val, value_proj_b, value, MMEM_, 256, 256, 0);
  gemm(qb16, w_samp, samp_off_b, offb, T_, 256, 256, 0);
  gemm(qb16, w_attn, attn_w_b, awb, T_, 128, 256, 0);
  aw_softmax<<<(T_ * NH_ + 255) / 256, 256, 0, stream>>>(awb);
  deform_attn<<<(T_ * 256 + 255) / 256, 256, 0, stream>>>(offb, awb, ref, value, dout16);
  gemm(dout16, w_oproj, output_proj_b, ca, T_, 256, 256, 0);
  resid_ln<<<(T_ + 3) / 4, 128, 0, stream>>>(t2, ca, n1g, n1b, t3, t3_16, T_);

  // --- FFN ---
  gemm(t3_16, w_lin1, lin1_b, ffh16, T_, 1024, 256, GEMM_RELU | GEMM_OUTF16);
  gemm(ffh16, w_lin2, lin2_b, ffo, T_, 256, 1024, 0);
  resid_ln<<<(T_ + 3) / 4, 128, 0, stream>>>(t3, ffo, n3g, n3b, outp,
                                             (_Float16*)nullptr, T_);
}
